// MultiHeadedAttention_65446711656945
// MI455X (gfx1250) — matrix operand order repaired and checked by reading, not compiled, not run on hardware
//
#include <hip/hip_runtime.h>
#include <math.h>

typedef __attribute__((ext_vector_type(16))) __bf16 v16bf;
typedef __attribute__((ext_vector_type(8)))  __bf16 v8bf;
typedef __attribute__((ext_vector_type(8)))  float  v8f;
typedef __attribute__((ext_vector_type(16))) float  v16f;

#define D_MODEL 256
#define KBLK    32
#define KB_PITCH 264   // ushorts per K row in LDS: 256 + 8 pad -> 528 B (16B aligned, bank-stride 4)
#define VT_PITCH 40    // ushorts per Vt row in LDS: 32 + 8 pad  -> 80 B  (16B aligned, bank-stride 20)

// Native conversions: let the backend emit (packed) v_cvt bf16 ops instead of
// integer bit-twiddling.
static __device__ __forceinline__ __bf16 f2bf(float x) { return (__bf16)x; }

static __device__ __forceinline__ v8bf cvt8(v8f f) {
  return __builtin_convertvector(f, v8bf);
}
static __device__ __forceinline__ v16bf cvt16(v16f f) {
  return __builtin_convertvector(f, v16bf);
}

static __device__ __forceinline__ v16bf combine8(v8bf lo, v8bf hi) {
  v16bf r;
#pragma unroll
  for (int i = 0; i < 8; ++i) { r[i] = lo[i]; r[8 + i] = hi[i]; }
  return r;
}

static __device__ __forceinline__ v16f pack16(float4 f0, float4 f1, float4 f2, float4 f3) {
  v16f v;
  v[0]=f0.x; v[1]=f0.y; v[2] =f0.z; v[3] =f0.w;
  v[4]=f1.x; v[5]=f1.y; v[6] =f1.z; v[7] =f1.w;
  v[8]=f2.x; v[9]=f2.y; v[10]=f2.z; v[11]=f2.w;
  v[12]=f3.x; v[13]=f3.y; v[14]=f3.z; v[15]=f3.w;
  return v;
}

// ---------------------------------------------------------------------------
// Kernel 1: streaming (flash) attention.  Each wave owns 16 query rows and
// iterates over keys in blocks of 32, keeping the 16x256 f32 output
// accumulator in registers.  S^T = K_blk * Q^T via WMMA (A = K rows,
// B = Q columns); softmax stats are per-lane + one shfl_xor(16); the exp'd
// probabilities are already in A-operand layout for P*V.
// ---------------------------------------------------------------------------
__global__ __launch_bounds__(128) void attn_fwd_kernel(
    const float* __restrict__ Qg, const float* __restrict__ Kg,
    const float* __restrict__ Vg, float* __restrict__ Xg, int N) {
  __shared__ __align__(16) unsigned short Kb[KBLK * KB_PITCH];       // 16.5 KB
  __shared__ __align__(16) unsigned short Vt[D_MODEL * VT_PITCH];    // 20.0 KB

  const int tid   = threadIdx.x;
  const int lane  = tid & 31;
  const int l16   = lane & 15;
  const int lhalf = (lane >> 4) & 1;          // 0: lanes 0-15, 1: lanes 16-31
  const int wave  = tid >> 5;
  const int qrow0 = blockIdx.x * 64 + wave * 16;

  // --- Q resident in registers as WMMA B-operand (query = lane column) ---
  // B layout: lanes 0-15 hold d-index c*32+0..15, lanes 16-31 hold c*32+16..31
  v16bf qreg[8];
  {
    const float* qp = Qg + (size_t)(qrow0 + l16) * D_MODEL + (lhalf ? 16 : 0);
#pragma unroll
    for (int c = 0; c < 8; ++c) {
      const float4* p = (const float4*)(qp + c * 32);
      qreg[c] = cvt16(pack16(p[0], p[1], p[2], p[3]));
    }
  }

  const v8f vzero = {0.f, 0.f, 0.f, 0.f, 0.f, 0.f, 0.f, 0.f};
  v8f o[16];                                   // 16 queries x 256 d (f32 accum)
#pragma unroll
  for (int i = 0; i < 16; ++i) o[i] = vzero;
  float m = -INFINITY;                         // running max (scaled log2 domain)
  float l = 0.f;                               // running denom
  const float SC = 0.0625f * 1.44269504088896340736f;  // 1/sqrt(256) * log2(e)

  for (int kb = 0; kb < N; kb += KBLK) {
    __syncthreads();
    // ---- stage K block (32 x 256) row-major bf16 into LDS, cooperative ----
#pragma unroll
    for (int i = 0; i < 8; ++i) {
      int g = tid + i * 128;                   // 0..1023 groups of 8 floats
      int key = g >> 5, c8 = g & 31;
      const float4* p = (const float4*)(Kg + (size_t)(kb + key) * D_MODEL + c8 * 8);
      float4 f0 = p[0], f1 = p[1];
      v8f f; f[0]=f0.x; f[1]=f0.y; f[2]=f0.z; f[3]=f0.w;
             f[4]=f1.x; f[5]=f1.y; f[6]=f1.z; f[7]=f1.w;
      *(v8bf*)(&Kb[key * KB_PITCH + c8 * 8]) = cvt8(f);
    }
    // ---- stage V block transposed: Vt[d][key], via 8x8 register tiles ----
    {
      int d8 = tid >> 2, k8 = tid & 3;         // 32 d-groups x 4 key-groups
      float tile[8][8];
#pragma unroll
      for (int j = 0; j < 8; ++j) {
        const float4* p = (const float4*)(Vg + (size_t)(kb + k8 * 8 + j) * D_MODEL + d8 * 8);
        float4 f0 = p[0], f1 = p[1];
        tile[j][0]=f0.x; tile[j][1]=f0.y; tile[j][2]=f0.z; tile[j][3]=f0.w;
        tile[j][4]=f1.x; tile[j][5]=f1.y; tile[j][6]=f1.z; tile[j][7]=f1.w;
      }
#pragma unroll
      for (int dd = 0; dd < 8; ++dd) {
        v8f f;
#pragma unroll
        for (int j = 0; j < 8; ++j) f[j] = tile[j][dd];
        *(v8bf*)(&Vt[(d8 * 8 + dd) * VT_PITCH + k8 * 8]) = cvt8(f);
      }
    }
    __syncthreads();

    // ---- S^T = K_blk * Q^T : two 16-key sub-tiles, K-dim 256 = 8 chunks ----
    v8f s0 = vzero, s1 = vzero;
#pragma unroll
    for (int c = 0; c < 8; ++c) {
      int aoff = c * 32 + (lhalf ? 16 : 0);    // A layout d-split per lane half
      v16bf a0 = combine8(*(const v8bf*)&Kb[l16 * KB_PITCH + aoff],
                          *(const v8bf*)&Kb[l16 * KB_PITCH + aoff + 8]);
      v16bf a1 = combine8(*(const v8bf*)&Kb[(16 + l16) * KB_PITCH + aoff],
                          *(const v8bf*)&Kb[(16 + l16) * KB_PITCH + aoff + 8]);
      s0 = __builtin_amdgcn_wmma_f32_16x16x32_bf16(false, a0, false, qreg[c],
                                                   (short)0, s0, false, false);
      s1 = __builtin_amdgcn_wmma_f32_16x16x32_bf16(false, a1, false, qreg[c],
                                                   (short)0, s1, false, false);
    }

    // ---- online softmax (query = lane, 16 in-lane keys + xor-16 merge) ----
    float p0[8], p1[8];
    float mb = -INFINITY;
#pragma unroll
    for (int r = 0; r < 8; ++r) {
      float v0 = s0[r] * SC, v1 = s1[r] * SC;
      p0[r] = v0; p1[r] = v1;
      mb = fmaxf(mb, fmaxf(v0, v1));
    }
    mb = fmaxf(mb, __shfl_xor(mb, 16));
    float mnew  = fmaxf(m, mb);
    float alpha = __builtin_amdgcn_exp2f(m - mnew);  // 0 on first block (m=-inf)
    float lb = 0.f;
    v16bf pa;                                  // P in A-operand layout, free!
#pragma unroll
    for (int r = 0; r < 8; ++r) {
      float e0 = __builtin_amdgcn_exp2f(p0[r] - mnew);
      float e1 = __builtin_amdgcn_exp2f(p1[r] - mnew);
      lb += e0 + e1;
      pa[r] = f2bf(e0); pa[8 + r] = f2bf(e1);
    }
    lb += __shfl_xor(lb, 16);
    l = l * alpha + lb;

    // rescale accumulator only when some query's max moved (wave-uniform
    // branch; expected only ~O(log #blocks) times per wave)
    if (__any(mnew > m)) {
      float av[8];
#pragma unroll
      for (int r = 0; r < 8; ++r) av[r] = __shfl(alpha, (lhalf ? 8 : 0) + r);
#pragma unroll
      for (int ch = 0; ch < 16; ++ch) {
#pragma unroll
        for (int r = 0; r < 8; ++r) o[ch][r] *= av[r];
      }
    }
    m = mnew;

    // ---- O += P * V : B operand = Vt column slice (contiguous in LDS) ----
#pragma unroll
    for (int ch = 0; ch < 16; ++ch) {
      int d = ch * 16 + l16;
      int koff = lhalf ? 8 : 0;                // keys 0-7,16-23 vs 8-15,24-31
      v16bf vb = combine8(*(const v8bf*)&Vt[d * VT_PITCH + koff],
                          *(const v8bf*)&Vt[d * VT_PITCH + koff + 16]);
      o[ch] = __builtin_amdgcn_wmma_f32_16x16x32_bf16(false, pa, false, vb,
                                                      (short)0, o[ch], false, false);
    }
  }

  // ---- normalize and write x to workspace ----
  float linv = 1.f / l;
  float lv[8];
#pragma unroll
  for (int r = 0; r < 8; ++r) lv[r] = __shfl(linv, (lhalf ? 8 : 0) + r);
  float* xr = Xg + (size_t)(qrow0 + (lhalf ? 8 : 0)) * D_MODEL + l16;
#pragma unroll
  for (int ch = 0; ch < 16; ++ch) {
#pragma unroll
    for (int r = 0; r < 8; ++r)
      xr[(size_t)r * D_MODEL + ch * 16] = o[ch][r] * lv[r];
  }
}

// ---------------------------------------------------------------------------
// Kernel 2: output projection y = x @ W^T + b via WMMA.
// A = x tile (16 q x 32 d), B = W^T chunk (rows of W read contiguously),
// D layout gives e-column = lane, query = component.
// ---------------------------------------------------------------------------
__global__ __launch_bounds__(128) void out_proj_kernel(
    const float* __restrict__ Xg, const float* __restrict__ Wg,
    const float* __restrict__ Bg, float* __restrict__ Yg, int N) {
  const int tid   = threadIdx.x;
  const int lane  = tid & 31;
  const int l16   = lane & 15;
  const int lhalf = (lane >> 4) & 1;
  const int wave  = tid >> 5;
  const int qrow0 = blockIdx.x * 64 + wave * 16;

  // A operand: lane<16 -> d = c*32 + {0..7,16..23}; lane>=16 -> {8..15,24..31}
  v16bf ax[8];
  {
    const float* xp = Xg + (size_t)(qrow0 + l16) * D_MODEL + (lhalf ? 8 : 0);
#pragma unroll
    for (int c = 0; c < 8; ++c) {
      const float4* p0 = (const float4*)(xp + c * 32);
      const float4* p1 = (const float4*)(xp + c * 32 + 16);
      ax[c] = cvt16(pack16(p0[0], p0[1], p1[0], p1[1]));
    }
  }

  const v8f vzero = {0.f, 0.f, 0.f, 0.f, 0.f, 0.f, 0.f, 0.f};
#pragma unroll 1
  for (int ech = 0; ech < 16; ++ech) {
    int e = ech * 16 + l16;
    float bias = Bg[e];
    const float* wp = Wg + (size_t)e * D_MODEL + (lhalf ? 8 : 0);
    v8f acc = vzero;
#pragma unroll
    for (int c = 0; c < 8; ++c) {
      const float4* p = (const float4*)(wp + c * 32);
      v16bf wb = cvt16(pack16(p[0], p[1], p[4], p[5]));
      acc = __builtin_amdgcn_wmma_f32_16x16x32_bf16(false, ax[c], false, wb,
                                                    (short)0, acc, false, false);
    }
    float* yr = Yg + (size_t)(qrow0 + (lhalf ? 8 : 0)) * D_MODEL + e;
#pragma unroll
    for (int r = 0; r < 8; ++r) yr[(size_t)r * D_MODEL] = acc[r] + bias;
  }
}

extern "C" void kernel_launch(void* const* d_in, const int* in_sizes, int n_in,
                              void* d_out, int out_size, void* d_ws, size_t ws_size,
                              hipStream_t stream) {
  const float* Q = (const float*)d_in[0];
  const float* K = (const float*)d_in[1];
  const float* V = (const float*)d_in[2];
  const float* W = (const float*)d_in[3];
  const float* B = (const float*)d_in[4];
  float* Y = (float*)d_out;
  float* X = (float*)d_ws;                     // [N, 256] f32 intermediate

  int N = in_sizes[0] / D_MODEL;               // 8192
  dim3 blk(128);                               // 4 waves, 64 query rows / WG
  dim3 grd(N / 64);                            // 128 workgroups
  attn_fwd_kernel<<<grd, blk, 0, stream>>>(Q, K, V, X, N);
  out_proj_kernel<<<grd, blk, 0, stream>>>(X, W, B, Y, N);
}
